// LSTMDecoder_70317204570321
// MI455X (gfx1250) — matrix operand order repaired and checked by reading, not compiled, not run on hardware
//
#include <hip/hip_runtime.h>

// ---------------------------------------------------------------------------
// LSTM greedy decoder for MI455X (gfx1250), wave32 + WMMA bf16 + async-to-LDS.
// H=128, V=32000, CD=256, T=21, NS=5, B=5120.
// ---------------------------------------------------------------------------

#define B_TOT   5120
#define H_DIM   128
#define V_DIM   32000
#define CD_DIM  256
#define T_STEPS 21
#define SOS_TOK 1
#define EOS_TOK 2

typedef __attribute__((ext_vector_type(16))) __bf16 v16bf;
typedef __attribute__((ext_vector_type(8)))  __bf16 v8bf;
typedef __attribute__((ext_vector_type(8)))  float  v8f;
typedef __attribute__((ext_vector_type(4)))  float  v4f;
typedef __attribute__((ext_vector_type(4)))  int    v4i;

#if __has_builtin(__builtin_amdgcn_global_load_async_to_lds_b128)
#define HAVE_ASYNC_LDS 1
#else
#define HAVE_ASYNC_LDS 0
#endif

// ---- async copy helpers ----------------------------------------------------
__device__ __forceinline__ void copy16_g2l(const void* g, void* l) {
#if HAVE_ASYNC_LDS
    // builtin expects (int4 addrspace(1)*, int4 addrspace(3)*, imm offset, imm cpol)
    __builtin_amdgcn_global_load_async_to_lds_b128(
        (__attribute__((address_space(1))) v4i*)g,
        (__attribute__((address_space(3))) v4i*)l, 0, 0);
#else
    *(int4*)l = *(const int4*)g;            // sync fallback: global->vgpr->lds
#endif
}

template <int N>
__device__ __forceinline__ void wait_async_le() {
#if HAVE_ASYNC_LDS && __has_builtin(__builtin_amdgcn_s_wait_asynccnt)
    __builtin_amdgcn_s_wait_asynccnt(N);
#endif
}

// ---- transcendental helpers (v_exp_f32 / v_log_f32) -----------------------
__device__ __forceinline__ float fast_exp(float x) {
    return __builtin_amdgcn_exp2f(x * 1.44269504088896340736f);
}
__device__ __forceinline__ float fast_log(float x) {
    return __builtin_amdgcn_logf(x) * 0.69314718055994530942f;
}
__device__ __forceinline__ float sigmoidf_(float x) {
    return 1.0f / (1.0f + fast_exp(-x));
}
__device__ __forceinline__ float tanhf_(float x) {
    float e = fast_exp(2.0f * x);
    return (e - 1.0f) / (e + 1.0f);
}

// ---- WMMA fragment loaders -------------------------------------------------
// A fragment (16x32 bf16): lane holds row (lane&15); half hh=lane>>4 gets
// K = k0+hh*8..+7 (elems 0..7) and K = k0+16+hh*8..+7 (elems 8..15).
__device__ __forceinline__ v16bf frag_a_f32(const float* rowbase, int k0, int hh) {
    const float* p = rowbase + k0 + hh * 8;
    v4f a0 = *(const v4f*)(p);
    v4f a1 = *(const v4f*)(p + 4);
    v4f b0 = *(const v4f*)(p + 16);
    v4f b1 = *(const v4f*)(p + 20);
    v16bf f;
    f[0]  = (__bf16)a0[0]; f[1]  = (__bf16)a0[1]; f[2]  = (__bf16)a0[2]; f[3]  = (__bf16)a0[3];
    f[4]  = (__bf16)a1[0]; f[5]  = (__bf16)a1[1]; f[6]  = (__bf16)a1[2]; f[7]  = (__bf16)a1[3];
    f[8]  = (__bf16)b0[0]; f[9]  = (__bf16)b0[1]; f[10] = (__bf16)b0[2]; f[11] = (__bf16)b0[3];
    f[12] = (__bf16)b1[0]; f[13] = (__bf16)b1[1]; f[14] = (__bf16)b1[2]; f[15] = (__bf16)b1[3];
    return f;
}

__device__ __forceinline__ v16bf frag_a_bf16(const __bf16* rowbase, int k0, int hh) {
    const __bf16* p = rowbase + k0 + hh * 8;
    v8bf lo = *(const v8bf*)(p);        // K offsets 0..7
    v8bf hi = *(const v8bf*)(p + 16);   // K offsets 16..23
    return __builtin_shufflevector(lo, hi, 0,1,2,3,4,5,6,7,8,9,10,11,12,13,14,15);
}

// B fragment (32x16 bf16): lane holds column n0+(lane&15); half hh=lane>>4
// gets K = k0+hh*8..+7 (elems 0..7) and K = k0+16+hh*8..+7 (elems 8..15). Source row-major [N][K].
__device__ __forceinline__ v16bf frag_b_bf16(const __bf16* W, int ldk, int n0, int k0, int lane) {
    int cn = n0 + (lane & 15);
    int hh = (lane >> 4) & 1;
    const __bf16* p = W + (size_t)cn * ldk + k0 + hh * 8;
    v8bf lo = *(const v8bf*)(p);
    v8bf hi = *(const v8bf*)(p + 16);
    return __builtin_shufflevector(lo, hi, 0,1,2,3,4,5,6,7,8,9,10,11,12,13,14,15);
}

__device__ __forceinline__ v8f wmma_bf16(v16bf a, v16bf b, v8f c) {
    return __builtin_amdgcn_wmma_f32_16x16x32_bf16(false, a, false, b, (short)0, c,
                                                   false, false);
}

// ---------------------------------------------------------------------------
// Kernel: init decode state
// ---------------------------------------------------------------------------
__global__ void init_state_kernel(int* __restrict__ it, int* __restrict__ unf) {
    int i = blockIdx.x * blockDim.x + threadIdx.x;
    if (i < B_TOT) { it[i] = SOS_TOK; unf[i] = 1; }
}

// ---------------------------------------------------------------------------
// Kernel: generic f32 -> bf16 conversion (weights are static: convert once)
// ---------------------------------------------------------------------------
__global__ void cvt_bf16_kernel(const float* __restrict__ src, __bf16* __restrict__ dst, int n) {
    for (int i = blockIdx.x * blockDim.x + threadIdx.x; i < n; i += gridDim.x * blockDim.x)
        dst[i] = (__bf16)src[i];
}

// ---------------------------------------------------------------------------
// Kernel: fuse [W_ih ; W_hh] -> Wg (512 x 256, bf16) so the gates GEMM is one
// K=256 stream of B fragments.
// ---------------------------------------------------------------------------
__global__ void build_wg_kernel(const float* __restrict__ Wih, const float* __restrict__ Whh,
                                __bf16* __restrict__ Wg) {
    int i = blockIdx.x * blockDim.x + threadIdx.x;
    if (i < 512 * 256) {
        int n = i >> 8, k = i & 255;
        float v = (k < 128) ? Wih[n * 128 + k] : Whh[n * 128 + (k - 128)];
        Wg[i] = (__bf16)v;
    }
}

// ---------------------------------------------------------------------------
// Kernel: fc  — hc = repeat(x,5) @ W_fc.T + b_fc ;  h0/c0 split.
// 320 blocks x 256 thr; block = 16 rows; wave w -> cols w*16 (+0/+128).
// ---------------------------------------------------------------------------
__global__ __launch_bounds__(256) void fc_kernel(const float* __restrict__ xin,
                                                 const __bf16* __restrict__ Wfcb,
                                                 const float* __restrict__ bfc,
                                                 float* __restrict__ h,
                                                 float* __restrict__ c) {
    const int tid  = threadIdx.x;
    const int lane = tid & 31, w = tid >> 5;
    const int r = lane & 15, hh = lane >> 4;
    const int m0 = blockIdx.x * 16;
    const float* rowx = xin + (size_t)((m0 + r) / 5) * CD_DIM;

    v16bf A[8];
#pragma unroll
    for (int kb = 0; kb < 8; ++kb) A[kb] = frag_a_f32(rowx, kb * 32, hh);

#pragma unroll
    for (int half = 0; half < 2; ++half) {
        int n0 = w * 16 + half * 128;
        v8f acc = {0.f,0.f,0.f,0.f,0.f,0.f,0.f,0.f};
#pragma unroll
        for (int kb = 0; kb < 8; ++kb) {
            v16bf Bf = frag_b_bf16(Wfcb, CD_DIM, n0, kb * 32, lane);
            acc = wmma_bf16(A[kb], Bf, acc);
        }
        int col = n0 + r;
        float bias = bfc[col];
#pragma unroll
        for (int i = 0; i < 8; ++i) {
            int mrow = m0 + hh * 8 + i;
            float v = acc[i] + bias;
            if (half == 0) h[(size_t)mrow * H_DIM + col] = v;
            else           c[(size_t)mrow * H_DIM + (col - 128)] = v;
        }
    }
}

// ---------------------------------------------------------------------------
// Kernel: gates + LSTM cell (fused).
// gates = [embed[it], h] (Bx256) @ Wg.T (256x512) + b_ih + b_hh ; cell update.
// 160 blocks x 256 thr; block = 32 rows; waves: rowtile = w>>2, stripe = w&3.
// ---------------------------------------------------------------------------
__global__ __launch_bounds__(256) void gates_cell_kernel(const __bf16* __restrict__ embb,
                                                         const __bf16* __restrict__ Wg,
                                                         const float* __restrict__ b_ih,
                                                         const float* __restrict__ b_hh,
                                                         float* __restrict__ h,
                                                         float* __restrict__ c,
                                                         const int* __restrict__ it) {
    __shared__ float gbuf[32][512];                    // 64 KB of 320 KB WGP LDS
    const int tid  = threadIdx.x;
    const int lane = tid & 31, w = tid >> 5;
    const int rt = w >> 2, cw = w & 3;
    const int r = lane & 15, hh = lane >> 4;
    const int m0 = blockIdx.x * 32;
    const int m  = m0 + rt * 16 + r;

    const __bf16* rowe = embb + (size_t)it[m] * H_DIM; // token-embedding gather
    const float*  rowh = h + (size_t)m * H_DIM;

    v16bf A[8];
#pragma unroll
    for (int kb = 0; kb < 4; ++kb) A[kb]     = frag_a_bf16(rowe, kb * 32, hh);
#pragma unroll
    for (int kb = 0; kb < 4; ++kb) A[4 + kb] = frag_a_f32(rowh, kb * 32, hh);

    for (int j = 0; j < 8; ++j) {
        int n0 = cw * 16 + j * 64;
        v8f acc = {0.f,0.f,0.f,0.f,0.f,0.f,0.f,0.f};
#pragma unroll
        for (int kb = 0; kb < 8; ++kb) {
            v16bf Bf = frag_b_bf16(Wg, CD_DIM, n0, kb * 32, lane);
            acc = wmma_bf16(A[kb], Bf, acc);
        }
        int col = n0 + r;
        float bias = b_ih[col] + b_hh[col];
#pragma unroll
        for (int i = 0; i < 8; ++i)
            gbuf[rt * 16 + hh * 8 + i][col] = acc[i] + bias;
    }
    __syncthreads();

    for (int e = tid; e < 32 * H_DIM; e += 256) {      // fused LSTM cell
        int ml = e >> 7, jc = e & 127;
        int gm = m0 + ml;
        float gi = gbuf[ml][jc];
        float gf = gbuf[ml][128 + jc];
        float gg = gbuf[ml][256 + jc];
        float go = gbuf[ml][384 + jc];
        float co = c[(size_t)gm * H_DIM + jc];
        float cn = sigmoidf_(gf) * co + sigmoidf_(gi) * tanhf_(gg);
        float hn = sigmoidf_(go) * tanhf_(cn);
        c[(size_t)gm * H_DIM + jc] = cn;
        h[(size_t)gm * H_DIM + jc] = hn;
    }
}

// ---------------------------------------------------------------------------
// Kernel: h1 = h @ W1.T + b1   (5120x128x128)
// ---------------------------------------------------------------------------
__global__ __launch_bounds__(256) void h1_kernel(const float* __restrict__ h,
                                                 const __bf16* __restrict__ W1b,
                                                 const float* __restrict__ b1,
                                                 float* __restrict__ h1) {
    const int tid  = threadIdx.x;
    const int lane = tid & 31, w = tid >> 5;
    const int r = lane & 15, hh = lane >> 4;
    const int m0 = blockIdx.x * 16;
    const float* rowh = h + (size_t)(m0 + r) * H_DIM;

    v16bf A[4];
#pragma unroll
    for (int kb = 0; kb < 4; ++kb) A[kb] = frag_a_f32(rowh, kb * 32, hh);

    int n0 = w * 16;
    v8f acc = {0.f,0.f,0.f,0.f,0.f,0.f,0.f,0.f};
#pragma unroll
    for (int kb = 0; kb < 4; ++kb) {
        v16bf Bf = frag_b_bf16(W1b, H_DIM, n0, kb * 32, lane);
        acc = wmma_bf16(A[kb], Bf, acc);
    }
    int col = n0 + r;
    float bias = b1[col];
#pragma unroll
    for (int i = 0; i < 8; ++i)
        h1[(size_t)(m0 + hh * 8 + i) * H_DIM + col] = acc[i] + bias;
}

// ---------------------------------------------------------------------------
// Kernel: vocab projection + fused argmax/log-softmax + greedy state update.
// W2 tile per iteration is one contiguous 16 KB chunk (64 rows x 256 B) ->
// double-buffered async load into LDS (ASYNCcnt), WMMA B-fragments from LDS.
// 160 blocks x 256 thr; block = 32 rows; waves: rowtile = w>>2, stripe = w&3.
// ---------------------------------------------------------------------------
__global__ __launch_bounds__(256) void logits_kernel(const float* __restrict__ h1,
                                                     const __bf16* __restrict__ w2b,
                                                     const float* __restrict__ b2,
                                                     int* __restrict__ it,
                                                     int* __restrict__ unf,
                                                     float* __restrict__ out_seq,
                                                     float* __restrict__ out_lp,
                                                     float* __restrict__ out_mask,
                                                     int t) {
    __shared__ __bf16 smem[2][64 * H_DIM];             // 2 x 16 KB W2 tiles
    __shared__ float red_m[2][16][64];
    __shared__ float red_s[2][16][64];
    __shared__ int   red_a[2][16][64];

    const int tid  = threadIdx.x;
    const int lane = tid & 31, w = tid >> 5;
    const int rt = w >> 2, cw = w & 3;
    const int r = lane & 15, hh = lane >> 4;
    const int m0 = blockIdx.x * 32;
    const int NITER = V_DIM / 64;                      // 500

    const char* gbase = (const char*)w2b;

    // cooperative tile fill: 16384 B / 256 thr = 64 B/thread = 4 x b128
    auto issue_tile = [&](int jj, int buf) {
        const char* g = gbase + (size_t)jj * 16384 + tid * 64;
        char* l = (char*)&smem[buf][0] + tid * 64;
#pragma unroll
        for (int i = 0; i < 4; ++i)
            copy16_g2l(g + 16 * i, l + 16 * i);
    };

    const float* rowh1 = h1 + (size_t)(m0 + rt * 16 + r) * H_DIM;
    v16bf A[4];
#pragma unroll
    for (int kb = 0; kb < 4; ++kb) A[kb] = frag_a_f32(rowh1, kb * 32, hh);

    float mx[8], sx[8];
    int   ax[8];
#pragma unroll
    for (int i = 0; i < 8; ++i) { mx[i] = -3.0e38f; sx[i] = 0.0f; ax[i] = 0; }

    issue_tile(0, 0);                                  // prologue

    for (int j = 0; j < NITER; ++j) {
        int buf = j & 1;
        if (j + 1 < NITER) {
            issue_tile(j + 1, buf ^ 1);
            wait_async_le<4>();                        // tile j complete
        } else {
            wait_async_le<0>();
        }
        __syncthreads();                               // tile j visible to all

        v8f acc = {0.f,0.f,0.f,0.f,0.f,0.f,0.f,0.f};
        const __bf16* tile = &smem[buf][0];
#pragma unroll
        for (int kb = 0; kb < 4; ++kb) {
            v16bf Bf = frag_b_bf16(tile, H_DIM, cw * 16, kb * 32, lane);  // ds_load
            acc = wmma_bf16(A[kb], Bf, acc);
        }

        int col = j * 64 + cw * 16 + r;
        float bias = b2[col];
#pragma unroll
        for (int i = 0; i < 8; ++i) {
            float x = acc[i] + bias;                   // logit
            if (x > mx[i]) { mx[i] = x; ax[i] = col; } // first-index on ties
            sx[i] += fast_exp(x);                      // logits O(+-5): safe
        }
        __syncthreads();                               // buf reusable at j+2
    }

    int cid = cw * 16 + r;                             // 64 contributors / row
#pragma unroll
    for (int i = 0; i < 8; ++i) {
        int row16 = hh * 8 + i;
        red_m[rt][row16][cid] = mx[i];
        red_s[rt][row16][cid] = sx[i];
        red_a[rt][row16][cid] = ax[i];
    }
    __syncthreads();

    if (tid < 32) {                                    // one lane per row
        int rt2 = tid >> 4, row = tid & 15;
        float bm = -3.0e38f, S = 0.0f;
        int   ba = 0;
        for (int cidq = 0; cidq < 64; ++cidq) {
            float mm = red_m[rt2][row][cidq];
            int   aa = red_a[rt2][row][cidq];
            S += red_s[rt2][row][cidq];
            if (mm > bm || (mm == bm && aa < ba)) { bm = mm; ba = aa; }
        }
        int gb = m0 + rt2 * 16 + row;
        int u  = unf[gb];
        float lp = bm - fast_log(S);                   // log_softmax at argmax
        int seq  = u ? ba : 0;
        size_t o = (size_t)gb * T_STEPS + t;
        out_seq[o]  = (float)seq;
        out_lp[o]   = lp;
        out_mask[o] = (float)u;
        it[gb]  = seq;
        unf[gb] = (u && seq != EOS_TOK) ? 1 : 0;
    }
}

// ---------------------------------------------------------------------------
// Host entry
// ---------------------------------------------------------------------------
extern "C" void kernel_launch(void* const* d_in, const int* in_sizes, int n_in,
                              void* d_out, int out_size, void* d_ws, size_t ws_size,
                              hipStream_t stream) {
    (void)in_sizes; (void)n_in; (void)out_size; (void)ws_size;

    const float* xin   = (const float*)d_in[0];   // input_features (1024, 256)
    const float* embed = (const float*)d_in[1];   // (V, H)
    const float* W_fc  = (const float*)d_in[2];   // (256, 256)
    const float* b_fc  = (const float*)d_in[3];
    const float* W_ih  = (const float*)d_in[4];   // (512, 128)
    const float* b_ih  = (const float*)d_in[5];
    const float* W_hh  = (const float*)d_in[6];   // (512, 128)
    const float* b_hh  = (const float*)d_in[7];
    const float* W1    = (const float*)d_in[8];   // (128, 128)
    const float* b1    = (const float*)d_in[9];
    const float* W2    = (const float*)d_in[10];  // (32000, 128)
    const float* b2    = (const float*)d_in[11];

    // workspace layout
    char*  ws    = (char*)d_ws;
    float* hbuf  = (float*)ws;                               // B*H f32
    float* cbuf  = hbuf + (size_t)B_TOT * H_DIM;             // B*H f32
    float* h1buf = cbuf + (size_t)B_TOT * H_DIM;             // B*H f32
    __bf16* w2b  = (__bf16*)(h1buf + (size_t)B_TOT * H_DIM); // V*H bf16
    __bf16* embb = w2b + (size_t)V_DIM * H_DIM;              // V*H bf16
    __bf16* wgb  = embb + (size_t)V_DIM * H_DIM;             // 512*256 bf16
    __bf16* w1b  = wgb + 512 * 256;                          // 128*128 bf16
    __bf16* wfcb = w1b + 128 * 128;                          // 256*256 bf16
    int* itbuf   = (int*)(wfcb + 256 * 256);                 // B
    int* unfbuf  = itbuf + B_TOT;                            // B

    float* out      = (float*)d_out;
    float* out_seq  = out;
    float* out_lp   = out + (size_t)B_TOT * T_STEPS;
    float* out_mask = out + 2 * (size_t)B_TOT * T_STEPS;

    init_state_kernel<<<(B_TOT + 255) / 256, 256, 0, stream>>>(itbuf, unfbuf);
    cvt_bf16_kernel<<<2048, 256, 0, stream>>>(W2, w2b, V_DIM * H_DIM);
    cvt_bf16_kernel<<<2048, 256, 0, stream>>>(embed, embb, V_DIM * H_DIM);
    cvt_bf16_kernel<<<64,   256, 0, stream>>>(W1, w1b, 128 * 128);
    cvt_bf16_kernel<<<256,  256, 0, stream>>>(W_fc, wfcb, 256 * 256);
    build_wg_kernel<<<512,  256, 0, stream>>>(W_ih, W_hh, wgb);

    fc_kernel<<<B_TOT / 16, 256, 0, stream>>>(xin, wfcb, b_fc, hbuf, cbuf);

    for (int t = 0; t < T_STEPS; ++t) {
        gates_cell_kernel<<<B_TOT / 32, 256, 0, stream>>>(embb, wgb, b_ih, b_hh,
                                                          hbuf, cbuf, itbuf);
        h1_kernel<<<B_TOT / 16, 256, 0, stream>>>(hbuf, w1b, b1, h1buf);
        logits_kernel<<<B_TOT / 32, 256, 0, stream>>>(h1buf, w2b, b2, itbuf, unfbuf,
                                                      out_seq, out_lp, out_mask, t);
    }
}
